// ContextQueryAttention_47347719471306
// MI455X (gfx1250) — matrix operand order repaired and checked by reading, not compiled, not run on hardware
//
#include <hip/hip_runtime.h>
#include <stdint.h>

#define NV 64
#define NW 128
#define DD 512
#define CN_PITCH 520   // 128 rows * 520 u16  (padded: 260 dwords -> +4 banks/row)
#define QN_PITCH 520   //  64 rows * 520 u16
#define QT_PITCH 72    // 512 rows *  72 u16  (36 dwords -> +36 banks/row)
#define P_PITCH  72    // 128 rows *  72 u16

typedef __attribute__((ext_vector_type(16))) __bf16 v16bf;
typedef __attribute__((ext_vector_type(8)))  float  v8f;

union ABu { uint4 u[2]; v16bf v; };

static __device__ __forceinline__ unsigned short f2bf(float f) {
  union { float f; unsigned int u; } c; c.f = f;
  unsigned int u = c.u;
  u += 0x7FFFu + ((u >> 16) & 1u);      // round-to-nearest-even
  return (unsigned short)(u >> 16);
}

__global__ __launch_bounds__(256, 1)
void cqattn_kernel(const float* __restrict__ qry,
                   const float* __restrict__ qmask,
                   const float* __restrict__ ctx,
                   const float* __restrict__ cmask,
                   float* __restrict__ out)
{
  extern __shared__ __align__(16) unsigned short smem[];
  unsigned short* cn = smem;                    // normalized context, bf16
  unsigned short* qn = cn + NW * CN_PITCH;      // normalized query,  bf16
  unsigned short* qT = qn + NV * QN_PITCH;      // raw query transposed [d][v], bf16
  unsigned short* pP = qT + DD * QT_PITCH;      // probabilities [w][v], bf16

  const int bc   = blockIdx.x;                  // b*nc + c
  const int lane = threadIdx.x & 31;
  const int wid  = threadIdx.x >> 5;            // 0..7 (8 waves)
  const int h    = lane >> 4;                   // lane half
  const int l16  = lane & 15;

  const float* qbase = qry + (size_t)bc * NV * DD;
  const float* cbase = ctx + (size_t)bc * NW * DD;

  // ---------------- Phase 0: L2 normalize + stage bf16 tiles ----------------
  // context: 16 rows per wave
  for (int r = 0; r < 16; ++r) {
    const int row = wid * 16 + r;
    const float4* src = (const float4*)(cbase + (size_t)row * DD);
    float4 v[4]; float ss = 0.f;
    #pragma unroll
    for (int j = 0; j < 4; ++j) {
      v[j] = src[j * 32 + lane];
      ss += v[j].x*v[j].x + v[j].y*v[j].y + v[j].z*v[j].z + v[j].w*v[j].w;
    }
    #pragma unroll
    for (int m = 1; m < 32; m <<= 1) ss += __shfl_xor(ss, m, 32);
    const float rn = 1.0f / fmaxf(sqrtf(ss), 1e-12f);
    #pragma unroll
    for (int j = 0; j < 4; ++j) {
      uint2 pk;
      pk.x = (unsigned)f2bf(v[j].x*rn) | ((unsigned)f2bf(v[j].y*rn) << 16);
      pk.y = (unsigned)f2bf(v[j].z*rn) | ((unsigned)f2bf(v[j].w*rn) << 16);
      *(uint2*)&cn[row * CN_PITCH + j * 128 + lane * 4] = pk;
    }
  }
  // query: 8 rows per wave -> qn (normalized) and qT (raw, transposed)
  for (int r = 0; r < 8; ++r) {
    const int row = wid * 8 + r;
    const float4* src = (const float4*)(qbase + (size_t)row * DD);
    float4 v[4]; float ss = 0.f;
    #pragma unroll
    for (int j = 0; j < 4; ++j) {
      v[j] = src[j * 32 + lane];
      ss += v[j].x*v[j].x + v[j].y*v[j].y + v[j].z*v[j].z + v[j].w*v[j].w;
    }
    #pragma unroll
    for (int m = 1; m < 32; m <<= 1) ss += __shfl_xor(ss, m, 32);
    const float rn = 1.0f / fmaxf(sqrtf(ss), 1e-12f);
    #pragma unroll
    for (int j = 0; j < 4; ++j) {
      uint2 pk;
      pk.x = (unsigned)f2bf(v[j].x*rn) | ((unsigned)f2bf(v[j].y*rn) << 16);
      pk.y = (unsigned)f2bf(v[j].z*rn) | ((unsigned)f2bf(v[j].w*rn) << 16);
      *(uint2*)&qn[row * QN_PITCH + j * 128 + lane * 4] = pk;
      const int d0 = j * 128 + lane * 4;
      qT[(d0 + 0) * QT_PITCH + row] = f2bf(v[j].x);
      qT[(d0 + 1) * QT_PITCH + row] = f2bf(v[j].y);
      qT[(d0 + 2) * QT_PITCH + row] = f2bf(v[j].z);
      qT[(d0 + 3) * QT_PITCH + row] = f2bf(v[j].w);
    }
  }
  __syncthreads();

  // ---------------- Phase 1: S = Cn * Qn^T  (per-wave 16x64 strip) ----------
  const int w0 = wid * 16;
  v8f acc[4];
  #pragma unroll
  for (int vt = 0; vt < 4; ++vt) acc[vt] = (v8f){0.f,0.f,0.f,0.f,0.f,0.f,0.f,0.f};

  const int arow = w0 + l16;                 // A-operand row per lane
  for (int kt = 0; kt < 16; ++kt) {          // K = 512 in steps of 32
    const int k0 = kt * 32 + 8 * h;
    ABu a;
    a.u[0] = *(const uint4*)&cn[arow * CN_PITCH + k0];
    a.u[1] = *(const uint4*)&cn[arow * CN_PITCH + k0 + 16];
    const int kb = kt * 32 + 8 * h;
    #pragma unroll
    for (int vt = 0; vt < 4; ++vt) {
      const int brow = vt * 16 + l16;        // B-operand column (= query row)
      ABu b;
      b.u[0] = *(const uint4*)&qn[brow * QN_PITCH + kb];
      b.u[1] = *(const uint4*)&qn[brow * QN_PITCH + kb + 16];
      acc[vt] = __builtin_amdgcn_wmma_f32_16x16x32_bf16(
          false, a.v, false, b.v, (short)0, acc[vt], false, false);
    }
  }

  // ---------------- Softmax over v (rows live in-wave) ----------------------
  const float scale = 0.044194173824159216f;   // 1/sqrt(512)
  float cmv[8], qmv[4], s[4][8], rmax[8], rsum[8];
  #pragma unroll
  for (int r = 0; r < 8; ++r) cmv[r] = cmask[(size_t)bc * NW + w0 + 8 * h + r];
  #pragma unroll
  for (int vt = 0; vt < 4; ++vt) qmv[vt] = qmask[(size_t)bc * NV + vt * 16 + l16];

  #pragma unroll
  for (int vt = 0; vt < 4; ++vt)
    #pragma unroll
    for (int r = 0; r < 8; ++r) {
      float mm = cmv[r] * qmv[vt];
      s[vt][r] = acc[vt][r] * scale - 1e10f * (1.0f - mm);
    }
  #pragma unroll
  for (int r = 0; r < 8; ++r) {
    float m = fmaxf(fmaxf(s[0][r], s[1][r]), fmaxf(s[2][r], s[3][r]));
    #pragma unroll
    for (int x = 1; x < 16; x <<= 1) m = fmaxf(m, __shfl_xor(m, x, 32));
    rmax[r] = m;
  }
  #pragma unroll
  for (int r = 0; r < 8; ++r) {
    float acc_s = 0.f;
    #pragma unroll
    for (int vt = 0; vt < 4; ++vt) {
      float e = __expf(s[vt][r] - rmax[r]);
      s[vt][r] = e; acc_s += e;
    }
    #pragma unroll
    for (int x = 1; x < 16; x <<= 1) acc_s += __shfl_xor(acc_s, x, 32);
    rsum[r] = 1.0f / acc_s;
  }
  #pragma unroll
  for (int vt = 0; vt < 4; ++vt)
    #pragma unroll
    for (int r = 0; r < 8; ++r) {
      float pv = s[vt][r] * rsum[r] * cmv[r] * qmv[vt];
      pP[(w0 + r + 8 * h) * P_PITCH + vt * 16 + l16] = f2bf(pv);
    }

  // ---------------- Phase 2: Out = P * Q  (K = 64, B from qT) ---------------
  ABu a0, a1;
  {
    const int prow = w0 + l16;
    const int k0 = 8 * h;
    a0.u[0] = *(const uint4*)&pP[prow * P_PITCH + k0];
    a0.u[1] = *(const uint4*)&pP[prow * P_PITCH + k0 + 16];
    a1.u[0] = *(const uint4*)&pP[prow * P_PITCH + 32 + k0];
    a1.u[1] = *(const uint4*)&pP[prow * P_PITCH + 32 + k0 + 16];
  }
  float* obase = out + (size_t)bc * NW * DD;
  const int kb2 = 8 * h;
  for (int dt = 0; dt < 32; ++dt) {
    const int drow = dt * 16 + l16;          // qT row = d column
    v8f o = (v8f){0.f,0.f,0.f,0.f,0.f,0.f,0.f,0.f};
    ABu b;
    b.u[0] = *(const uint4*)&qT[drow * QT_PITCH + kb2];
    b.u[1] = *(const uint4*)&qT[drow * QT_PITCH + kb2 + 16];
    o = __builtin_amdgcn_wmma_f32_16x16x32_bf16(false, a0.v, false, b.v, (short)0, o, false, false);
    b.u[0] = *(const uint4*)&qT[drow * QT_PITCH + 32 + kb2];
    b.u[1] = *(const uint4*)&qT[drow * QT_PITCH + 32 + kb2 + 16];
    o = __builtin_amdgcn_wmma_f32_16x16x32_bf16(false, a1.v, false, b.v, (short)0, o, false, false);
    #pragma unroll
    for (int r = 0; r < 8; ++r)
      obase[(size_t)(w0 + r + 8 * h) * DD + dt * 16 + l16] = o[r];
  }
}

extern "C" void kernel_launch(void* const* d_in, const int* in_sizes, int n_in,
                              void* d_out, int out_size, void* d_ws, size_t ws_size,
                              hipStream_t stream) {
  const float* qry   = (const float*)d_in[0];
  const float* qmask = (const float*)d_in[1];
  const float* ctx   = (const float*)d_in[2];
  const float* cmask = (const float*)d_in[3];
  float* out = (float*)d_out;

  const int nbc = in_sizes[3] / NW;  // bs * nc  (context_mask flat count / nw)
  const size_t smem_bytes =
      (size_t)(NW * CN_PITCH + NV * QN_PITCH + DD * QT_PITCH + NW * P_PITCH) *
      sizeof(unsigned short);        // ~292 KB of the 320 KB WGP LDS

  hipFuncSetAttribute((const void*)cqattn_kernel,
                      hipFuncAttributeMaxDynamicSharedMemorySize,
                      (int)smem_bytes);
  cqattn_kernel<<<nbc, 256, smem_bytes, stream>>>(qry, qmask, ctx, cmask, out);
}
